// SentenceAttNet_24017457119662
// MI455X (gfx1250) — matrix operand order repaired and checked by reading, not compiled, not run on hardware
//
#include <hip/hip_runtime.h>
#include <hip/hip_bf16.h>

#define SEQ   256
#define BATCH 4096
#define DIM   100           // D
#define NPAD  112           // N padded to 7 * 16
#define KPAD  128           // K padded to 4 * 32
#define ROWS  (SEQ * BATCH) // 1,048,576 rows of the big GEMM
#define HN    (2 * BATCH * (DIM / 2)) // h_output elements = 409600

typedef __attribute__((ext_vector_type(16))) _Float16 v16h;
typedef __attribute__((ext_vector_type(8)))  _Float16 v8h;
typedef __attribute__((ext_vector_type(8)))  float    v8f;

// Branchless fast tanh: tanh(x) = 1 - 2/(exp2(2*log2e*x) + 1).
// One v_exp_f32 + one v_rcp_f32 + two FMAs; saturates to +/-1 for large |x|.
__device__ __forceinline__ float fast_tanh(float x) {
    float e = __builtin_amdgcn_exp2f(x * 2.88539008177792681f); // 2*log2(e)
    float r = __builtin_amdgcn_rcpf(e + 1.0f);
    return __builtin_fmaf(-2.0f, r, 1.0f);
}

// ---------------------------------------------------------------------------
// Kernel 0: convert sentence_weight [100x100] f32 -> padded, transposed f16
// buffer wh[NPAD][KPAD] laid out exactly as the WMMA B-fragment LDS image:
// wh[n*KPAD + k] = W[k][n], zero padded.
// ---------------------------------------------------------------------------
__global__ __launch_bounds__(256) void pack_w_kernel(const float* __restrict__ W,
                                                     _Float16* __restrict__ wh) {
    int i = blockIdx.x * 256 + threadIdx.x;   // i < NPAD*KPAD = 14336
    int n = i >> 7;          // / KPAD
    int k = i & (KPAD - 1);  // % KPAD
    float v = (n < DIM && k < DIM) ? W[k * DIM + n] : 0.0f;
    wh[i] = (_Float16)v;
}

// ---------------------------------------------------------------------------
// Kernel 1: scores[r] = tanh( tanh(f[r,:] @ W + bias) . cw )  for all rows r.
// 128 threads = 4 waves; each wave owns a 16-row tile -> 64 rows per block.
// ---------------------------------------------------------------------------
__global__ __launch_bounds__(128) void attn_scores_kernel(
    const float*    __restrict__ f,      // [ROWS][DIM]
    const _Float16* __restrict__ wh,     // [NPAD][KPAD] f16, B-fragment layout
    const float*    __restrict__ bias,   // [DIM]
    const float*    __restrict__ cw,     // [DIM]
    float*          __restrict__ scores) // [ROWS]
{
    __shared__ __align__(32) _Float16 w_lds[NPAD * KPAD];   // 28 KB
    __shared__ __align__(16) _Float16 a_lds[4][16 * KPAD];  // 4 x 4 KB
    __shared__ float bias_lds[NPAD];
    __shared__ float cw_lds[NPAD];

    const int tid  = threadIdx.x;
    const int lane = tid & 31;
    const int wave = tid >> 5;
    const int mcol = lane & 15;          // n / m position within 16
    const int hi   = (lane >> 4) & 1;    // lane group: 0 = lanes 0-15, 1 = 16-31

    // ---- Stage W into LDS via CDNA5 async global->LDS DMA (ASYNCcnt path) ----
    {
        const int nvec = (NPAD * KPAD * (int)sizeof(_Float16)) / 16; // 1792 x b128
        for (int i = tid; i < nvec; i += 128) {
            unsigned long long gaddr =
                (unsigned long long)(const void*)((const char*)wh + i * 16);
            unsigned ldsoff = (unsigned)(unsigned long long)(void*)
                              ((char*)w_lds + i * 16);
            asm volatile("global_load_async_to_lds_b128 %0, %1, off"
                         :: "v"(ldsoff), "v"(gaddr) : "memory");
        }
    }

    // ---- bias / context weight (zero padded) ----
    if (tid < NPAD) {
        bias_lds[tid] = (tid < DIM) ? bias[tid] : 0.0f;
        cw_lds[tid]   = (tid < DIM) ? cw[tid]   : 0.0f;
    }

    // ---- Stage this wave's 16-row A tile, fp32 -> f16, zero padded ----
    const long rowBase = (long)blockIdx.x * 64 + (long)wave * 16;
    _Float16* at = &a_lds[wave][0];
    for (int m = 0; m < 16; ++m) {
        const float* src = f + (rowBase + m) * DIM;
        for (int c = lane; c < KPAD; c += 32) {
            at[m * KPAD + c] = (c < DIM) ? (_Float16)src[c] : (_Float16)0.0f;
        }
    }

    asm volatile("s_wait_asynccnt 0" ::: "memory");   // W DMA done
    __syncthreads();

    // ---- Build A fragments for the 4 k-steps (contiguous k order, same as B) ----
    // lanes 0-15 : halves 0..15 = K kb+0..15
    // lanes 16-31: halves 0..15 = K kb+16..31
    v16h afrag[4];
#pragma unroll
    for (int kk = 0; kk < 4; ++kk) {
        int k0 = kk * 32 + hi * 16;
        v8h lo = *(const v8h*)&at[mcol * KPAD + k0];
        v8h hv = *(const v8h*)&at[mcol * KPAD + k0 + 8];
        afrag[kk] = __builtin_shufflevector(lo, hv, 0, 1, 2, 3, 4, 5, 6, 7,
                                                    8, 9, 10, 11, 12, 13, 14, 15);
    }

    // ---- 7 n-tiles x 4 k-steps of v_wmma_f32_16x16x32_f16, fused epilogue ----
    float contrib[8];
#pragma unroll
    for (int i = 0; i < 8; ++i) contrib[i] = 0.0f;

#pragma unroll
    for (int t = 0; t < 7; ++t) {
        v8f acc = {};
        const int n = t * 16 + mcol;
#pragma unroll
        for (int kk = 0; kk < 4; ++kk) {
            // B fragment: column n per lane; halves = K kb..kb+15
            const int kb = kk * 32 + hi * 16;
            v16h bfrag = *(const v16h*)&w_lds[n * KPAD + kb];
            acc = __builtin_amdgcn_wmma_f32_16x16x32_f16(
                /*neg_a=*/false, afrag[kk], /*neg_b=*/false, bfrag,
                /*c_mod=*/(short)0, acc, /*reuse_a=*/false, /*reuse_b=*/false);
        }
        const float bb = bias_lds[n];
        const float cc = cw_lds[n];
#pragma unroll
        for (int i = 0; i < 8; ++i)
            contrib[i] += fast_tanh(acc[i] + bb) * cc;  // row m = i + 8*hi
    }

    // ---- reduce each row's dot product across the 16 lanes of its group ----
#pragma unroll
    for (int i = 0; i < 8; ++i) {
        float v = contrib[i];
        v += __shfl_xor(v, 1, 32);
        v += __shfl_xor(v, 2, 32);
        v += __shfl_xor(v, 4, 32);
        v += __shfl_xor(v, 8, 32);
        contrib[i] = v;
    }
    if (mcol == 0) {
        long r = rowBase + hi * 8;
#pragma unroll
        for (int i = 0; i < 8; ++i)
            scores[r + i] = fast_tanh(contrib[i]);
    }
}

// ---------------------------------------------------------------------------
// Kernel 2: per batch column b: softmax over seq, then pooled[b,d] =
// sum_s attn[s] * f[s,b,d]. One 256-thread block per batch element.
// ---------------------------------------------------------------------------
__global__ __launch_bounds__(256) void attn_pool_kernel(
    const float* __restrict__ f,       // [SEQ][BATCH][DIM]
    const float* __restrict__ scores,  // [SEQ][BATCH] (row r = s*BATCH+b)
    float*       __restrict__ out)     // pooled [BATCH][DIM]
{
    __shared__ float sc[SEQ];
    __shared__ float red[256];
    __shared__ float comb[128];

    const int b   = blockIdx.x;
    const int tid = threadIdx.x;

    float v = scores[(long)tid * BATCH + b];   // blockDim == SEQ == 256
    sc[tid]  = v;
    red[tid] = v;
    __syncthreads();
    for (int s2 = 128; s2 > 0; s2 >>= 1) {
        if (tid < s2) red[tid] = fmaxf(red[tid], red[tid + s2]);
        __syncthreads();
    }
    const float mx = red[0];
    __syncthreads();
    const float e = __builtin_amdgcn_exp2f((v - mx) * 1.44269504088896341f);
    red[tid] = e;
    __syncthreads();
    for (int s2 = 128; s2 > 0; s2 >>= 1) {
        if (tid < s2) red[tid] += red[tid + s2];
        __syncthreads();
    }
    const float inv = 1.0f / red[0];
    sc[tid] = e * inv;                 // attention weight for seq position tid
    __syncthreads();

    // weighted sum over seq: two thread-halves each take odd/even s.
    const int half = tid >> 7;         // 0 or 1
    const int d    = tid & 127;
    float acc = 0.0f;
    if (d < DIM) {
        const long stride = (long)BATCH * DIM;
        const float* fp = f + (long)half * stride + (long)b * DIM + d;
#pragma unroll 4
        for (int s = half; s < SEQ; s += 2) {
            acc += sc[s] * (*fp);
            fp += 2 * stride;
        }
    }
    if (half == 1 && d < DIM) comb[d] = acc;
    __syncthreads();
    if (half == 0 && d < DIM) out[(long)b * DIM + d] = acc + comb[d];
}

// ---------------------------------------------------------------------------
// Kernel 3: pass h_output straight through to d_out[409600:]
// ---------------------------------------------------------------------------
__global__ __launch_bounds__(256) void copy_h_kernel(const float* __restrict__ src,
                                                     float* __restrict__ dst, int n) {
    int i = blockIdx.x * 256 + threadIdx.x;
    if (i < n) dst[i] = src[i];
}

extern "C" void kernel_launch(void* const* d_in, const int* in_sizes, int n_in,
                              void* d_out, int out_size, void* d_ws, size_t ws_size,
                              hipStream_t stream) {
    const float* f    = (const float*)d_in[0];  // [256,4096,100]
    const float* h    = (const float*)d_in[1];  // [2,4096,50]
    const float* W    = (const float*)d_in[2];  // [100,100]
    const float* bias = (const float*)d_in[3];  // [1,100]
    const float* cw   = (const float*)d_in[4];  // [100,1]
    float* out = (float*)d_out;

    // workspace layout: [0, ROWS*4) scores | then packed f16 W image
    float*    scores = (float*)d_ws;
    _Float16* wh     = (_Float16*)((char*)d_ws + (size_t)ROWS * sizeof(float));

    pack_w_kernel<<<(NPAD * KPAD) / 256, 256, 0, stream>>>(W, wh);
    attn_scores_kernel<<<ROWS / 64, 128, 0, stream>>>(f, wh, bias, cw, scores);
    attn_pool_kernel<<<BATCH, 256, 0, stream>>>(f, scores, out);
    copy_h_kernel<<<(HN + 255) / 256, 256, 0, stream>>>(h, out + (long)BATCH * DIM, HN);
}
